// DynamicDecoder_50946902065327
// MI455X (gfx1250) — matrix operand order repaired and checked by reading, not compiled, not run on hardware
//
#include <hip/hip_runtime.h>
#include <hip/hip_bf16.h>

typedef __attribute__((ext_vector_type(16))) __bf16         v16bf;
typedef __attribute__((ext_vector_type(8)))  float          v8f;
typedef __attribute__((ext_vector_type(16))) unsigned short v16u16;
typedef __attribute__((ext_vector_type(8)))  unsigned short v8u16;
typedef __attribute__((ext_vector_type(8)))  unsigned int   v8u32;

#define A_DIM   256   // D_IN
#define H_DIM   50
#define NPAD    64    // padded hidden (4 N-tiles of 16)
#define BLK     256   // threads per block (8 waves)
#define WAVES   8
#define MT_PER_WAVE 4 // 4 sequential 16-row M-tiles per wave -> 512 agents/block

// two f32 -> packed bf16x2 in one dword ([15:0]=lo, [31:16]=hi); RNE in HW
__device__ __forceinline__ unsigned int f2b2(float lo, float hi) {
#if defined(__gfx1250__)
  unsigned int r;
  asm("v_cvt_pk_bf16_f32 %0, %1, %2" : "=v"(r) : "v"(lo), "v"(hi));
  return r;
#else
  unsigned int ul = __float_as_uint(lo), uh = __float_as_uint(hi);
  ul = (ul + 0x7FFFu + ((ul >> 16) & 1u)) >> 16;
  uh = (uh + 0x7FFFu + ((uh >> 16) & 1u)) & 0xFFFF0000u;
  return uh | ul;
#endif
}

__device__ __forceinline__ unsigned short f2b(float f) {
  return (unsigned short)(f2b2(f, 0.0f) & 0xFFFFu);
}

// hardware tanh (TRANS op); trailing v_nop satisfies the TRANS hazard rule
__device__ __forceinline__ float fast_tanh(float x) {
#if defined(__gfx1250__)
  float r;
  asm("v_tanh_f32 %0, %1\n\tv_nop" : "=v"(r) : "v"(x));
  return r;
#else
  return tanhf(x);
#endif
}

// stable softplus on hardware exp/log TRANS ops
__device__ __forceinline__ float softplus_f(float x) {
  return fmaxf(x, 0.0f) + __logf(1.0f + __expf(-fabsf(x)));
}

// load 16 contiguous bf16 (as u16) from LDS
__device__ __forceinline__ v16u16 ld16(const unsigned short* p) {
  v8u16 lo = *(const v8u16*)(p);
  v8u16 hi = *(const v8u16*)(p + 8);
  v16u16 c;
#pragma unroll
  for (int i = 0; i < 8; ++i) { c[i] = lo[i]; c[i + 8] = hi[i]; }
  return c;
}

// load two separated 8-element chunks (A-fragment pattern)
__device__ __forceinline__ v16u16 ld8x2(const unsigned short* p0, const unsigned short* p1) {
  v8u16 lo = *(const v8u16*)(p0);
  v8u16 hi = *(const v8u16*)(p1);
  v16u16 c;
#pragma unroll
  for (int i = 0; i < 8; ++i) { c[i] = lo[i]; c[i + 8] = hi[i]; }
  return c;
}

__device__ __forceinline__ v8f wmma_a32(v8u32 a, v16u16 b, v8f c) {
  return __builtin_amdgcn_wmma_f32_16x16x32_bf16(
      false, __builtin_bit_cast(v16bf, a),
      false, __builtin_bit_cast(v16bf, b),
      (short)0, c, false, false);
}
__device__ __forceinline__ v8f wmma_a16(v16u16 a, v16u16 b, v8f c) {
  return __builtin_amdgcn_wmma_f32_16x16x32_bf16(
      false, __builtin_bit_cast(v16bf, a),
      false, __builtin_bit_cast(v16bf, b),
      (short)0, c, false, false);
}

__global__ __launch_bounds__(BLK) void dyndec_kernel(
    const float* __restrict__ noise, const float* __restrict__ ctx,
    const float* __restrict__ xprev, const float* __restrict__ dxin,
    const float* __restrict__ W0, const float* __restrict__ b0,
    const float* __restrict__ W1, const float* __restrict__ b1,
    const float* __restrict__ Wp, const float* __restrict__ bp,
    float* __restrict__ out, int A) {

  // ---- LDS: transposed bf16 weights + biases + per-wave staging (63040 B) ----
  __shared__ alignas(32) unsigned short w0t[NPAD * A_DIM];   // [n][k] 32768 B
  __shared__ alignas(32) unsigned short w1t[NPAD * NPAD];    // [n][k]  8192 B
  __shared__ alignas(32) unsigned short wpt[16 * NPAD];      // [n][k]  2048 B
  __shared__ float b0s[NPAD], b1s[NPAD], bps[16];
  __shared__ alignas(32) unsigned short hstage[WAVES * 16 * NPAD]; // 16384 B
  __shared__ float pst[WAVES * 16 * 6];                            //  3072 B

  const int tid  = threadIdx.x;
  const int wave = tid >> 5;
  const int lane = tid & 31;
  const int lhalf = lane >> 4;
  const int lidx  = lane & 15;

  // ---- stage weights: f32 global -> bf16 (packed cvt), transposed, padded ----
  {
    unsigned int* w0d = (unsigned int*)w0t;
    for (int e = tid; e < NPAD * (A_DIM / 2); e += BLK) {
      int n = e >> 7, k = (e & 127) * 2;
      float lo = (n < H_DIM) ? W0[k * H_DIM + n] : 0.0f;
      float hi = (n < H_DIM) ? W0[(k + 1) * H_DIM + n] : 0.0f;
      w0d[e] = f2b2(lo, hi);
    }
    unsigned int* w1d = (unsigned int*)w1t;
    for (int e = tid; e < NPAD * (NPAD / 2); e += BLK) {
      int n = e >> 5, k = (e & 31) * 2;
      float lo = (n < H_DIM && k     < H_DIM) ? W1[k * H_DIM + n]       : 0.0f;
      float hi = (n < H_DIM && k + 1 < H_DIM) ? W1[(k + 1) * H_DIM + n] : 0.0f;
      w1d[e] = f2b2(lo, hi);
    }
    unsigned int* wpd = (unsigned int*)wpt;
    for (int e = tid; e < 16 * (NPAD / 2); e += BLK) {
      int n = e >> 5, k = (e & 31) * 2;
      float lo = (n < 6 && k     < H_DIM) ? Wp[k * 6 + n]       : 0.0f;
      float hi = (n < 6 && k + 1 < H_DIM) ? Wp[(k + 1) * 6 + n] : 0.0f;
      wpd[e] = f2b2(lo, hi);
    }
    for (int e = tid; e < NPAD; e += BLK) {
      b0s[e] = (e < H_DIM) ? b0[e] : 0.0f;
      b1s[e] = (e < H_DIM) ? b1[e] : 0.0f;
    }
    if (tid < 16) bps[tid] = (tid < 6) ? bp[tid] : 0.0f;
  }
  __syncthreads();

  unsigned short* hs = hstage + wave * 16 * NPAD;

  for (int mt = 0; mt < MT_PER_WAVE; ++mt) {
    const int base = blockIdx.x * (WAVES * MT_PER_WAVE * 16) + wave * (MT_PER_WAVE * 16) + mt * 16;
    int rowA = base + lidx;
    int rowc = (rowA < A) ? rowA : (A - 1);          // clamp: keep EXEC full for WMMA
    const float* crow = ctx + (size_t)rowc * A_DIM;
    if (mt < MT_PER_WAVE - 1) {                      // prefetch next tile's row
      int nr = base + 16 + lidx;
      __builtin_prefetch(ctx + (size_t)((nr < A) ? nr : (A - 1)) * A_DIM, 0, 0);
    }

    // ---- GEMM1 A-fragments: 8 K-steps of 32, f32 global -> packed bf16 regs ----
    v8u32 afr[8];
#pragma unroll
    for (int ks = 0; ks < 8; ++ks) {
      const float4* p0 = (const float4*)(crow + ks * 32 + lhalf * 8);
      const float4* p1 = (const float4*)(crow + ks * 32 + 16 + lhalf * 8);
      float4 f0 = p0[0], f1 = p0[1], f2 = p1[0], f3 = p1[1];
      v8u32 v;
      v[0] = f2b2(f0.x, f0.y); v[1] = f2b2(f0.z, f0.w);
      v[2] = f2b2(f1.x, f1.y); v[3] = f2b2(f1.z, f1.w);
      v[4] = f2b2(f2.x, f2.y); v[5] = f2b2(f2.z, f2.w);
      v[6] = f2b2(f3.x, f3.y); v[7] = f2b2(f3.z, f3.w);
      afr[ks] = v;
    }

    // ---- GEMM1: h = softplus(X @ W0 + b0); K-outer, 4 accumulators,
    //      B-fragments double-buffered so WMMAs never wait on their own loads ----
    {
      v8f acc[4];
#pragma unroll
      for (int nt = 0; nt < 4; ++nt) {
        float bias = b0s[lidx + nt * 16];
#pragma unroll
        for (int i = 0; i < 8; ++i) acc[nt][i] = bias;
      }
      v16u16 bcur[4];
#pragma unroll
      for (int nt = 0; nt < 4; ++nt)
        bcur[nt] = ld8x2(w0t + (lidx + nt * 16) * A_DIM + lhalf * 8, w0t + (lidx + nt * 16) * A_DIM + lhalf * 8 + 16);
#pragma unroll
      for (int ks = 0; ks < 8; ++ks) {
        v16u16 bnxt[4];
        if (ks < 7) {
#pragma unroll
          for (int nt = 0; nt < 4; ++nt)
            bnxt[nt] = ld8x2(w0t + (lidx + nt * 16) * A_DIM + lhalf * 8 + (ks + 1) * 32, w0t + (lidx + nt * 16) * A_DIM + lhalf * 8 + (ks + 1) * 32 + 16);
        }
#pragma unroll
        for (int nt = 0; nt < 4; ++nt)
          acc[nt] = wmma_a32(afr[ks], bcur[nt], acc[nt]);
#pragma unroll
        for (int nt = 0; nt < 4; ++nt) bcur[nt] = bnxt[nt];
      }
#pragma unroll
      for (int nt = 0; nt < 4; ++nt)
#pragma unroll
        for (int r = 0; r < 8; ++r)
          hs[(r + lhalf * 8) * NPAD + lidx + nt * 16] = f2b(softplus_f(acc[nt][r]));
    }

    // ---- GEMM2: h2 = tanh(h @ W1 + b1), K padded to 64; all B preloaded ----
    {
      v16u16 a2[2];
#pragma unroll
      for (int ks = 0; ks < 2; ++ks) {
        const unsigned short* hr = hs + lidx * NPAD + ks * 32 + lhalf * 8;
        a2[ks] = ld8x2(hr, hr + 16);
      }
      v16u16 bw[2][4];
#pragma unroll
      for (int ks = 0; ks < 2; ++ks)
#pragma unroll
        for (int nt = 0; nt < 4; ++nt)
          bw[ks][nt] = ld8x2(w1t + (lidx + nt * 16) * NPAD + lhalf * 8 + ks * 32, w1t + (lidx + nt * 16) * NPAD + lhalf * 8 + ks * 32 + 16);
      v8f acc[4];
#pragma unroll
      for (int nt = 0; nt < 4; ++nt) {
        float bias = b1s[lidx + nt * 16];
#pragma unroll
        for (int i = 0; i < 8; ++i) acc[nt][i] = bias;
      }
#pragma unroll
      for (int ks = 0; ks < 2; ++ks)
#pragma unroll
        for (int nt = 0; nt < 4; ++nt)
          acc[nt] = wmma_a16(a2[ks], bw[ks][nt], acc[nt]);
#pragma unroll
      for (int nt = 0; nt < 4; ++nt)
#pragma unroll
        for (int r = 0; r < 8; ++r)
          hs[(r + lhalf * 8) * NPAD + lidx + nt * 16] = f2b(fast_tanh(acc[nt][r]));
    }

    // ---- GEMM3: pred = h2 @ Wp + bp (single N-tile, 6 valid columns) ----
    {
      v16u16 a3[2], b3[2];
#pragma unroll
      for (int ks = 0; ks < 2; ++ks) {
        const unsigned short* hr = hs + lidx * NPAD + ks * 32 + lhalf * 8;
        a3[ks] = ld8x2(hr, hr + 16);
        b3[ks] = ld8x2(wpt + lidx * NPAD + lhalf * 8 + ks * 32, wpt + lidx * NPAD + lhalf * 8 + ks * 32 + 16);
      }
      float bias = bps[lidx];
      v8f acc;
#pragma unroll
      for (int i = 0; i < 8; ++i) acc[i] = bias;
#pragma unroll
      for (int ks = 0; ks < 2; ++ks)
        acc = wmma_a16(a3[ks], b3[ks], acc);
      if (lidx < 6) {
#pragma unroll
        for (int r = 0; r < 8; ++r)
          pst[wave * 96 + (r + lhalf * 8) * 6 + lidx] = acc[r];
      }
    }

    // ---- epilogue: one agent per lane (lanes 0..15) ----
    if (lane < 16) {
      int a = base + lane;
      if (a < A) {
        const float* pp = pst + wave * 96 + lane * 6;
        float p0 = pp[0], p1 = pp[1], p2 = pp[2], p3 = pp[3], p4 = pp[4], p5 = pp[5];
        float2 nz = *(const float2*)(noise + (size_t)a * 2);
        float2 xp = *(const float2*)(xprev + (size_t)a * 2);
        float2 dv = *(const float2*)(dxin  + (size_t)a * 2);
        float mu0 = xp.x + 0.5f * dv.x + p0;
        float mu1 = xp.y + 0.5f * dv.y + p1;
        float bb  = p3 + p4;
        float apd = p2 + p5;
        float amd = p2 - p5;
        float delta = sqrtf(amd * amd + bb * bb);
        float ec = __expf(delta);
        float ic = 1.0f / ec;
        float ch = 0.5f * (ec + ic);
        float sinhc = (delta < 1e-6f) ? 1.0f : (0.5f * (ec - ic) / delta);
        float e  = __expf(apd);
        float s00 = e * (ch + amd * sinhc);
        float s01 = e * (bb * sinhc);
        float s11 = e * (ch - amd * sinhc);
        float x0 = s00 * nz.x + s01 * nz.y + mu0;
        float x1 = s01 * nz.x + s11 * nz.y + mu1;
        *(float2*)(out + (size_t)a * 2)                 = make_float2(x0, x1);
        *(float2*)(out + (size_t)2 * A + (size_t)a * 2) = make_float2(mu0, mu1);
        float4 sgv = make_float4(s00, s01, s01, s11);
        *(float4*)(out + (size_t)4 * A + (size_t)a * 4) = sgv;
      }
    }
  }
}

extern "C" void kernel_launch(void* const* d_in, const int* in_sizes, int n_in,
                              void* d_out, int out_size, void* d_ws, size_t ws_size,
                              hipStream_t stream) {
  const float* noise = (const float*)d_in[0];
  const float* ctx   = (const float*)d_in[1];
  const float* xprev = (const float*)d_in[2];
  const float* dx    = (const float*)d_in[3];
  const float* W0    = (const float*)d_in[4];
  const float* b0    = (const float*)d_in[5];
  const float* W1    = (const float*)d_in[6];
  const float* b1    = (const float*)d_in[7];
  const float* Wp    = (const float*)d_in[8];
  const float* bp    = (const float*)d_in[9];
  float* out = (float*)d_out;
  const int A = in_sizes[0] / 2;
  const int agents_per_block = WAVES * MT_PER_WAVE * 16; // 512
  const int blocks = (A + agents_per_block - 1) / agents_per_block;
  dyndec_kernel<<<blocks, BLK, 0, stream>>>(noise, ctx, xprev, dx,
                                            W0, b0, W1, b1, Wp, bp, out, A);
}
